// FragGNNSmall_28398323761365
// MI455X (gfx1250) — matrix operand order repaired and checked by reading, not compiled, not run on hardware
//
#include <hip/hip_runtime.h>
#include <hip/hip_bf16.h>

// ---------------- constants from the reference ----------------
static constexpr int cN_ATOM  = 131072;
static constexpr int cN_FRAG  = 32768;
static constexpr int cN_EDGE  = 262144;
static constexpr int cN_FEDGE = 65536;
static constexpr int cN_MEM   = 196608;
static constexpr int cB       = 4096;
static constexpr int cH       = 256;
static constexpr int cIN_A    = 64;
static constexpr int cIN_F    = 32;
static constexpr int cIN_E    = 16;
static constexpr int cL       = 4;

typedef float v8f __attribute__((ext_vector_type(8)));
typedef __bf16 v16bf __attribute__((ext_vector_type(16)));
typedef unsigned short v16u16 __attribute__((ext_vector_type(16)));

// RNE f32 -> bf16 via bit ops (no __bf16 scalar arithmetic needed)
__device__ __forceinline__ unsigned short f32_to_bf16_rne(float f) {
  unsigned u = __float_as_uint(f);
  u += 0x7fffu + ((u >> 16) & 1u);
  return (unsigned short)(u >> 16);
}
__device__ __forceinline__ float bf16_to_f32(unsigned short h) {
  return __uint_as_float((unsigned)h << 16);
}

// -------- weight split+pack into ISA B-fragment layout (bf16 hi/lo) --------
// Element e of the v16bf B fragment lives at K = kblk*32 + 16*(e>>3) + 8*sel + (e&7),
// col = nt*16 + (lane&15), lane = sel*16 + (col&15).
// Packed address: (((nt*kblocks + kblk)*32 + lane)*16 + e)
__global__ void repack_w_k(const float* __restrict__ W, unsigned short* __restrict__ hi,
                           unsigned short* __restrict__ lo, int K, int N) {
  int i = blockIdx.x * 256 + threadIdx.x;      // i < K*N (always divisible by 256 here)
  int kk  = i / N;
  int col = i - kk * N;
  float f = W[i];
  unsigned short h = f32_to_bf16_rne(f);
  unsigned short l = f32_to_bf16_rne(f - bf16_to_f32(h));
  int kblk   = kk >> 5;
  int within = kk & 31;
  int sel = (within >> 3) & 1;
  int e   = (within & 7) | ((within >> 4) << 3);
  int lane = sel * 16 + (col & 15);
  int nt   = col >> 4;
  int kblocks = K >> 5;
  size_t dst = (((size_t)nt * kblocks + kblk) * 32 + lane) * 16 + e;
  hi[dst] = h;
  lo[dst] = l;
}

// ---------------- bf16x3 WMMA GEMM (fp32-accurate) ----------------
// C[M,N] = op( A[g(row),K] @ W[K,N] + bias ), one wave per 16x64 tile.
// A is fp32, split in-register into bf16 hi/lo; W pre-split/packed.
// acc += Ahi*Bhi + Alo*Bhi + Ahi*Blo  (f32 accumulate; Alo*Blo negligible)
// Optional row gather (gidx), optional scatter-mean accumulate (sidx + scnt).
// M % 16 == 0, N % 64 == 0, K % 32 == 0 (true for all call sites here).
__global__ __launch_bounds__(32)
void gemm_bf16x3(const float* __restrict__ A,
                 const unsigned short* __restrict__ Bhi,
                 const unsigned short* __restrict__ Blo,
                 const float* __restrict__ bias, float* __restrict__ C,
                 int M, int K, int N,
                 const int* __restrict__ gidx, const int* __restrict__ sidx,
                 const float* __restrict__ scnt, int relu) {
  const int mtile = blockIdx.x;
  const int nt0   = blockIdx.y * 4;              // first of 4 N-subtiles
  const int lane  = threadIdx.x;
  const int mrow  = lane & 15;
  const int sel   = lane >> 4;                   // 0 / 1 lane half
  const int row   = mtile * 16 + mrow;
  const int arow  = gidx ? gidx[row] : row;
  const float* Ap = A + (size_t)arow * K;
  const int kblocks = K >> 5;

  v8f acc[4] = {};

  for (int kb = 0; kb < K; kb += 32) {
    __builtin_prefetch(Ap + kb + 32, 0, 3);      // next A block -> global_prefetch_b8

    // A fragment: element e <-> K = kb + 16*(e>>3) + 8*sel + (e&7)
    const float4 a0 = *(const float4*)(Ap + kb + 8 * sel);
    const float4 a1 = *(const float4*)(Ap + kb + 8 * sel + 4);
    const float4 a2 = *(const float4*)(Ap + kb + 16 + 8 * sel);
    const float4 a3 = *(const float4*)(Ap + kb + 16 + 8 * sel + 4);
    float r[16] = {a0.x, a0.y, a0.z, a0.w, a1.x, a1.y, a1.z, a1.w,
                   a2.x, a2.y, a2.z, a2.w, a3.x, a3.y, a3.z, a3.w};
    v16u16 ahu, alu;
#pragma unroll
    for (int e = 0; e < 16; ++e) {
      unsigned short h = f32_to_bf16_rne(r[e]);
      ahu[e] = h;
      alu[e] = f32_to_bf16_rne(r[e] - bf16_to_f32(h));
    }
    const v16bf ahi = __builtin_bit_cast(v16bf, ahu);
    const v16bf alo = __builtin_bit_cast(v16bf, alu);

#pragma unroll
    for (int j = 0; j < 4; ++j) {
      const size_t bbase =
          (((size_t)(nt0 + j) * kblocks + (kb >> 5)) * 32 + lane) * 16;
      const v16bf bh = __builtin_bit_cast(v16bf, *(const v16u16*)(Bhi + bbase));
      const v16bf bl = __builtin_bit_cast(v16bf, *(const v16u16*)(Blo + bbase));
      acc[j] = __builtin_amdgcn_wmma_f32_16x16x32_bf16(
          false, ahi, false, bh, (short)0, acc[j], false, false);
      acc[j] = __builtin_amdgcn_wmma_f32_16x16x32_bf16(
          false, alo, false, bh, (short)0, acc[j], false, false);
      acc[j] = __builtin_amdgcn_wmma_f32_16x16x32_bf16(
          false, ahi, false, bl, (short)0, acc[j], false, false);
    }
  }

  const int rbase = sel * 8;                     // lanes 16-31 hold M = 8..15
#pragma unroll
  for (int j = 0; j < 4; ++j) {
    const int col = (nt0 + j) * 16 + (lane & 15);
    const float bsv = bias ? bias[col] : 0.f;
#pragma unroll
    for (int r8 = 0; r8 < 8; ++r8) {
      const int rr = mtile * 16 + rbase + r8;
      float v = acc[j][r8] + bsv;
      if (relu) v = fmaxf(v, 0.f);
      if (sidx) {                                // fused scatter-mean accumulate
        const int t = sidx[rr];
        const float sc = 1.f / fmaxf(scnt[t], 1.f);
        atomicAdd(&C[(size_t)t * N + col], v * sc);
      } else {
        C[(size_t)rr * N + col] = v;
      }
    }
  }
}

// ---------------- elementwise / scatter helpers ----------------
__global__ void fzero_k(float* p, size_t n) {
  size_t i = (size_t)blockIdx.x * blockDim.x + threadIdx.x;
  if (i < n) p[i] = 0.f;
}

__global__ void fadd_k(float* y, const float* x, size_t n) {
  size_t i = (size_t)blockIdx.x * blockDim.x + threadIdx.x;
  if (i < n) y[i] += x[i];
}

// agg[i] = (1+eps[l])*x[i] + agg[i]   (GIN/GINE combine, in place)
__global__ void gin_combine_k(const float* __restrict__ x, float* __restrict__ agg,
                              const float* __restrict__ epsArr, int l, size_t n) {
  size_t i = (size_t)blockIdx.x * blockDim.x + threadIdx.x;
  if (i < n) agg[i] = (1.f + epsArr[l]) * x[i] + agg[i];
}

// fused bond encoder + GINE message: agg[dst] += relu(edge_attr@bondW + bondb + a[src])
__global__ __launch_bounds__(256)
void gine_edge_k(const float* __restrict__ a, const float* __restrict__ edge_attr,
                 const float* __restrict__ bondW, const float* __restrict__ bondb,
                 const int* __restrict__ src, const int* __restrict__ dst,
                 float* __restrict__ agg) {
  __shared__ float ea[cIN_E];
  const int e = blockIdx.x;
  const int c = threadIdx.x;                     // 0..255 (channel)
  if (c < cIN_E) ea[c] = edge_attr[(size_t)e * cIN_E + c];
  __syncthreads();
  const int s = src[e], d = dst[e];
  float v = bondb[c];
#pragma unroll
  for (int k = 0; k < cIN_E; ++k) v += ea[k] * bondW[k * cH + c];
  v += a[(size_t)s * cH + c];
  v = fmaxf(v, 0.f);
  atomicAdd(&agg[(size_t)d * cH + c], v);
}

// agg[dst[e]] += f[src[e]]  over 256 channels (fragment GIN aggregation)
__global__ void seg_gather_add_k(const float* __restrict__ f, const int* __restrict__ src,
                                 const int* __restrict__ dst, float* __restrict__ agg, int nEdge) {
  size_t t = (size_t)blockIdx.x * blockDim.x + threadIdx.x;
  int e = (int)(t >> 8), c = (int)(t & 255);
  if (e >= nEdge) return;
  atomicAdd(&agg[(size_t)dst[e] * cH + c], f[(size_t)src[e] * cH + c]);
}

__global__ void count_idx_k(const int* __restrict__ idx, float* __restrict__ cnt, int n) {
  int i = blockIdx.x * blockDim.x + threadIdx.x;
  if (i < n) atomicAdd(&cnt[idx[i]], 1.f);
}

// per-column sum / sumsq for training-mode BatchNorm (chunked over rows)
__global__ void colstats_k(const float* __restrict__ X, float* __restrict__ sum,
                           float* __restrict__ sumsq, int M, int C) {
  int c = blockIdx.x * blockDim.x + threadIdx.x;
  if (c >= C) return;
  int chunk = (M + gridDim.y - 1) / gridDim.y;
  int r0 = blockIdx.y * chunk;
  int r1 = min(r0 + chunk, M);
  float s = 0.f, q = 0.f;
  for (int r = r0; r < r1; ++r) {
    float v = X[(size_t)r * C + c];
    s += v; q += v * v;
  }
  atomicAdd(&sum[c], s);
  atomicAdd(&sumsq[c], q);
}

// X = relu( (X-mean)*rsqrt(var+1e-5)*g + b )
__global__ void bn_relu_k(float* __restrict__ X, const float* __restrict__ sum,
                          const float* __restrict__ sumsq, const float* __restrict__ g,
                          const float* __restrict__ b, int M, int C) {
  size_t i = (size_t)blockIdx.x * blockDim.x + threadIdx.x;
  if (i >= (size_t)M * C) return;
  int c = (int)(i % C);
  float invM = 1.f / (float)M;
  float mean = sum[c] * invM;
  float var  = sumsq[c] * invM - mean * mean;
  float v = (X[i] - mean) * rsqrtf(var + 1e-5f) * g[c] + b[c];
  X[i] = fmaxf(v, 0.f);
}

// pooled[seg[r]] += X[r]  (256 channels)
__global__ void pool_sum_k(const float* __restrict__ X, const int* __restrict__ seg,
                           float* __restrict__ out, int n) {
  size_t t = (size_t)blockIdx.x * blockDim.x + threadIdx.x;
  int r = (int)(t >> 8), c = (int)(t & 255);
  if (r >= n) return;
  atomicAdd(&out[(size_t)seg[r] * cH + c], X[(size_t)r * cH + c]);
}

__global__ void pool_div_k(float* __restrict__ out, const float* __restrict__ cnt, int rows) {
  size_t t = (size_t)blockIdx.x * blockDim.x + threadIdx.x;
  int r = (int)(t >> 8), c = (int)(t & 255);
  if (r >= rows) return;
  out[(size_t)r * cH + c] /= fmaxf(cnt[r], 1.f);
}

// out[row] = dot(X[row,:], w[:,0]) + b2   (final 256 -> 1 projection)
__global__ __launch_bounds__(256)
void final_out_k(const float* __restrict__ X, const float* __restrict__ w,
                 const float* __restrict__ b2, float* __restrict__ out, int rows) {
  __shared__ float sh[256];
  int r = blockIdx.x, c = threadIdx.x;
  sh[c] = X[(size_t)r * cH + c] * w[c];
  __syncthreads();
  for (int s = 128; s > 0; s >>= 1) {
    if (c < s) sh[c] += sh[c + s];
    __syncthreads();
  }
  if (c == 0) out[r] = sh[0] + b2[0];
}

// ---------------- host orchestration ----------------
extern "C" void kernel_launch(void* const* d_in, const int* in_sizes, int n_in,
                              void* d_out, int out_size, void* d_ws, size_t ws_size,
                              hipStream_t stream) {
  const float* x          = (const float*)d_in[0];
  const float* fragments  = (const float*)d_in[1];
  const float* edge_attr  = (const float*)d_in[2];
  const float* atom_enc_W = (const float*)d_in[3];
  const float* atom_enc_b = (const float*)d_in[4];
  const float* frag_enc_W = (const float*)d_in[5];
  const float* frag_enc_b = (const float*)d_in[6];
  const float* bond_W     = (const float*)d_in[7];
  const float* bond_b     = (const float*)d_in[8];
  const float* gine_eps   = (const float*)d_in[9];
  const float* gine_W1    = (const float*)d_in[10];
  const float* gine_b1    = (const float*)d_in[11];
  const float* gine_bn_g  = (const float*)d_in[12];
  const float* gine_bn_b  = (const float*)d_in[13];
  const float* gine_W2    = (const float*)d_in[14];
  const float* gine_b2    = (const float*)d_in[15];
  const float* atom_bn_g  = (const float*)d_in[16];
  const float* atom_bn_b  = (const float*)d_in[17];
  const float* gin_eps    = (const float*)d_in[18];
  const float* gin_W1     = (const float*)d_in[19];
  const float* gin_b1     = (const float*)d_in[20];
  const float* gin_bn_g   = (const float*)d_in[21];
  const float* gin_bn_b   = (const float*)d_in[22];
  const float* gin_W2     = (const float*)d_in[23];
  const float* gin_b2     = (const float*)d_in[24];
  const float* frag_bn_g  = (const float*)d_in[25];
  const float* frag_bn_b  = (const float*)d_in[26];
  const float* a2f_W      = (const float*)d_in[27];
  const float* a2f_b      = (const float*)d_in[28];
  const float* f2a_W      = (const float*)d_in[29];
  const float* f2a_b      = (const float*)d_in[30];
  const float* fo_W1      = (const float*)d_in[31];
  const float* fo_b1      = (const float*)d_in[32];
  const float* fo_W2      = (const float*)d_in[33];
  const float* fo_b2      = (const float*)d_in[34];
  const float* ao_W1      = (const float*)d_in[35];
  const float* ao_b1      = (const float*)d_in[36];
  const float* ao_W2      = (const float*)d_in[37];
  const float* ao_b2      = (const float*)d_in[38];
  const float* out_W1     = (const float*)d_in[39];
  const float* out_b1     = (const float*)d_in[40];
  const float* out_W2     = (const float*)d_in[41];
  const float* out_b2     = (const float*)d_in[42];
  const int* edge_index   = (const int*)d_in[43];
  const int* frag_edge_index = (const int*)d_in[44];
  const int* batch        = (const int*)d_in[45];
  const int* frag_batch   = (const int*)d_in[46];
  const int* mem_atom     = (const int*)d_in[47];
  const int* mem_frag     = (const int*)d_in[48];
  float* out = (float*)d_out;

  const int* e_src  = edge_index;
  const int* e_dst  = edge_index + cN_EDGE;
  const int* fe_src = frag_edge_index;
  const int* fe_dst = frag_edge_index + cN_FEDGE;

  // ---- workspace layout (floats) ----
  float* ws = (float*)d_ws;
  size_t o = 0;
  float* A0   = ws + o; o += (size_t)cN_ATOM * cH;        // atom features
  float* A1   = ws + o; o += (size_t)cN_ATOM * 2 * cH;    // atom hidden (512)
  float* AGGA = ws + o; o += (size_t)cN_ATOM * cH;        // atom agg / h
  float* F0   = ws + o; o += (size_t)cN_FRAG * cH;        // frag features
  float* F1   = ws + o; o += (size_t)cN_FRAG * 2 * cH;    // frag hidden (512)
  float* AGGF = ws + o; o += (size_t)cN_FRAG * cH;
  float* CNT_F  = ws + o; o += cN_FRAG;                   // |mem -> frag|
  float* CNT_A  = ws + o; o += cN_ATOM;                   // |mem -> atom|
  float* CNT_B  = ws + o; o += cB;                        // atoms per graph
  float* CNT_FB = ws + o; o += cB;                        // frags per graph
  float* POOL_A = ws + o; o += (size_t)cB * cH;
  float* POOL_F = ws + o; o += (size_t)cB * cH;
  float* POOL_T = ws + o; o += (size_t)cB * cH;
  float* SUMC   = ws + o; o += 512;
  float* SQC    = ws + o; o += 512;
  // packed bf16 hi/lo weights (max K*N = 512*256 elements each)
  unsigned short* WPH = (unsigned short*)(ws + o); o += (512 * 256) / 2;
  unsigned short* WPL = (unsigned short*)(ws + o); o += (512 * 256) / 2;
  (void)ws_size; (void)in_sizes; (void)n_in; (void)out_size;

  auto zero = [&](float* p, size_t n) {
    fzero_k<<<dim3((unsigned)((n + 255) / 256)), dim3(256), 0, stream>>>(p, n);
  };
  auto gemm = [&](const float* A, const float* Wf, const float* bias, float* C,
                  int M, int K, int N, const int* gi, const int* si,
                  const float* sc, int relu) {
    repack_w_k<<<dim3((K * N) / 256), dim3(256), 0, stream>>>(Wf, WPH, WPL, K, N);
    gemm_bf16x3<<<dim3(M / 16, N / 64), dim3(32), 0, stream>>>(
        A, WPH, WPL, bias, C, M, K, N, gi, si, sc, relu);
  };
  auto bn = [&](float* X, int M, int C, const float* g, const float* b) {
    zero(SUMC, 512); zero(SQC, 512);
    colstats_k<<<dim3((C + 255) / 256, 64), dim3(256), 0, stream>>>(X, SUMC, SQC, M, C);
    size_t n = (size_t)M * C;
    bn_relu_k<<<dim3((unsigned)((n + 255) / 256)), dim3(256), 0, stream>>>(
        X, SUMC, SQC, g, b, M, C);
  };

  // ---- segment counts (mem edges + pooling) ----
  zero(CNT_F, cN_FRAG); zero(CNT_A, cN_ATOM); zero(CNT_B, cB); zero(CNT_FB, cB);
  count_idx_k<<<dim3(cN_MEM / 256), dim3(256), 0, stream>>>(mem_frag, CNT_F, cN_MEM);
  count_idx_k<<<dim3(cN_MEM / 256), dim3(256), 0, stream>>>(mem_atom, CNT_A, cN_MEM);
  count_idx_k<<<dim3(cN_ATOM / 256), dim3(256), 0, stream>>>(batch, CNT_B, cN_ATOM);
  count_idx_k<<<dim3(cN_FRAG / 256), dim3(256), 0, stream>>>(frag_batch, CNT_FB, cN_FRAG);

  // ---- encoders ----
  gemm(x, atom_enc_W, atom_enc_b, A0, cN_ATOM, cIN_A, cH, nullptr, nullptr, nullptr, 0);
  gemm(fragments, frag_enc_W, frag_enc_b, F0, cN_FRAG, cIN_F, cH, nullptr, nullptr, nullptr, 0);

  // ---- layers ----
  for (int l = 0; l < cL; ++l) {
    // GINE on atom graph: fused bond-encoder + message scatter
    zero(AGGA, (size_t)cN_ATOM * cH);
    gine_edge_k<<<dim3(cN_EDGE), dim3(256), 0, stream>>>(
        A0, edge_attr, bond_W + (size_t)l * cIN_E * cH, bond_b + (size_t)l * cH,
        e_src, e_dst, AGGA);
    gin_combine_k<<<dim3((cN_ATOM * cH) / 256), dim3(256), 0, stream>>>(
        A0, AGGA, gine_eps, l, (size_t)cN_ATOM * cH);
    gemm(AGGA, gine_W1 + (size_t)l * cH * 2 * cH, gine_b1 + (size_t)l * 2 * cH, A1,
         cN_ATOM, cH, 2 * cH, nullptr, nullptr, nullptr, 0);
    bn(A1, cN_ATOM, 2 * cH, gine_bn_g + (size_t)l * 2 * cH, gine_bn_b + (size_t)l * 2 * cH);
    gemm(A1, gine_W2 + (size_t)l * 2 * cH * cH, gine_b2 + (size_t)l * cH, A0,
         cN_ATOM, 2 * cH, cH, nullptr, nullptr, nullptr, 0);
    bn(A0, cN_ATOM, cH, atom_bn_g + (size_t)l * cH, atom_bn_b + (size_t)l * cH);

    // GIN on fragment graph
    zero(AGGF, (size_t)cN_FRAG * cH);
    seg_gather_add_k<<<dim3((cN_FEDGE * cH) / 256), dim3(256), 0, stream>>>(
        F0, fe_src, fe_dst, AGGF, cN_FEDGE);
    gin_combine_k<<<dim3((cN_FRAG * cH) / 256), dim3(256), 0, stream>>>(
        F0, AGGF, gin_eps, l, (size_t)cN_FRAG * cH);
    gemm(AGGF, gin_W1 + (size_t)l * cH * 2 * cH, gin_b1 + (size_t)l * 2 * cH, F1,
         cN_FRAG, cH, 2 * cH, nullptr, nullptr, nullptr, 0);
    bn(F1, cN_FRAG, 2 * cH, gin_bn_g + (size_t)l * 2 * cH, gin_bn_b + (size_t)l * 2 * cH);
    gemm(F1, gin_W2 + (size_t)l * 2 * cH * cH, gin_b2 + (size_t)l * cH, F0,
         cN_FRAG, 2 * cH, cH, nullptr, nullptr, nullptr, 0);
    bn(F0, cN_FRAG, cH, frag_bn_g + (size_t)l * cH, frag_bn_b + (size_t)l * cH);

    // inter message passing (fused gather-GEMM-relu-scatter-mean, accumulates in place)
    gemm(A0, a2f_W + (size_t)l * cH * cH, a2f_b + (size_t)l * cH, F0,
         cN_MEM, cH, cH, mem_atom, mem_frag, CNT_F, 1);
    gemm(F0, f2a_W + (size_t)l * cH * cH, f2a_b + (size_t)l * cH, A0,
         cN_MEM, cH, cH, mem_frag, mem_atom, CNT_A, 1);
  }

  // ---- readout: fragment path ----
  zero(POOL_F, (size_t)cB * cH);
  pool_sum_k<<<dim3((cN_FRAG * cH) / 256), dim3(256), 0, stream>>>(F0, frag_batch, POOL_F, cN_FRAG);
  pool_div_k<<<dim3((cB * cH) / 256), dim3(256), 0, stream>>>(POOL_F, CNT_FB, cB);
  gemm(POOL_F, fo_W1, fo_b1, POOL_T, cB, cH, cH, nullptr, nullptr, nullptr, 1);
  gemm(POOL_T, fo_W2, fo_b2, POOL_F, cB, cH, cH, nullptr, nullptr, nullptr, 1);

  // ---- readout: atom path ----
  zero(POOL_A, (size_t)cB * cH);
  pool_sum_k<<<dim3((cN_ATOM * cH) / 256), dim3(256), 0, stream>>>(A0, batch, POOL_A, cN_ATOM);
  pool_div_k<<<dim3((cB * cH) / 256), dim3(256), 0, stream>>>(POOL_A, CNT_B, cB);
  gemm(POOL_A, ao_W1, ao_b1, POOL_T, cB, cH, cH, nullptr, nullptr, nullptr, 1);
  gemm(POOL_T, ao_W2, ao_b2, POOL_A, cB, cH, cH, nullptr, nullptr, nullptr, 1);

  // ---- combine + output head ----
  fadd_k<<<dim3((cB * cH) / 256), dim3(256), 0, stream>>>(POOL_A, POOL_F, (size_t)cB * cH);
  gemm(POOL_A, out_W1, out_b1, POOL_T, cB, cH, cH, nullptr, nullptr, nullptr, 1);
  final_out_k<<<dim3(cB), dim3(256), 0, stream>>>(POOL_T, out_W2, out_b2, out, cB);
}
